// TemporalAttention_1752346657135
// MI455X (gfx1250) — matrix operand order repaired and checked by reading, not compiled, not run on hardware
//
#include <hip/hip_runtime.h>

#define N_ROWS 8192
#define FDIM   1024
#define HDIM   1024
#define NCHUNK 4                 // column chunks for pass1/pass2 parallelism
#define CHUNKC (N_ROWS / NCHUNK)
#define NTILE  (CHUNKC / 16)     // 16-column tiles per chunk
#define MBLK   32                // row-blocks for passes: N/256

typedef __attribute__((ext_vector_type(16))) __bf16 v16bf;
typedef __attribute__((ext_vector_type(8)))  float  v8f;
typedef __attribute__((address_space(3)))    unsigned lds_u32;

union FragBF { uint4 q[2]; unsigned u[8]; v16bf v; };

// ---- bf16 helpers (round-to-nearest-even) ----
__device__ __forceinline__ unsigned short f2bf(float f) {
  unsigned u = __float_as_uint(f);
  u += 0x7FFFu + ((u >> 16) & 1u);
  return (unsigned short)(u >> 16);
}
__device__ __forceinline__ unsigned pack2bf(float lo, float hi) {
  return (unsigned)f2bf(lo) | ((unsigned)f2bf(hi) << 16);
}
__device__ __forceinline__ v8f zero8() {
  v8f z;
#pragma unroll
  for (int i = 0; i < 8; ++i) z[i] = 0.0f;
  return z;
}

// cross-lane reductions within each 16-lane half of a wave32
__device__ __forceinline__ float rmax16(float x) {
#pragma unroll
  for (int m = 1; m < 16; m <<= 1) x = fmaxf(x, __shfl_xor(x, m, 32));
  return x;
}
__device__ __forceinline__ float rsum16(float x) {
#pragma unroll
  for (int m = 1; m < 16; m <<= 1) x += __shfl_xor(x, m, 32);
  return x;
}

// ---- CDNA5 async global->LDS copy (ASYNCcnt-tracked, no VGPR round trip) ----
__device__ __forceinline__ unsigned lds_offset_of(const unsigned* p) {
  return (unsigned)(unsigned long long)(lds_u32*)p;
}
// Copy one 32KB K tile (16 rows x 512 uints) with 256 threads: 8 x b128 each.
__device__ __forceinline__ void async_tile_load(const unsigned* __restrict__ gsrc,
                                                unsigned* __restrict__ lbase) {
  const int tid = threadIdx.x;
#pragma unroll
  for (int i = 0; i < 8; ++i) {
    const int idx4 = tid + i * 256;                  // uint4 index, 2048 total
    const unsigned* g = gsrc + idx4 * 4;
    const unsigned loff = lds_offset_of(lbase + idx4 * 4);
    asm volatile("global_load_async_to_lds_b128 %0, %1, off"
                 :: "v"(loff), "v"(g) : "memory");
  }
}
__device__ __forceinline__ void wait_async0() {
  asm volatile("s_wait_asynccnt 0x0" ::: "memory");
}

// =====================================================================
// Kernel 1: Q/K projection. C[n,h] = sum_f x[n,f]*W[f,h] + b[h], bf16 out.
// Block: 256 thr (8 waves). Wave -> 16(M) x 64(N) via 4 WMMA chains.
// Grid: (H/64, N/128, 2)  z=0 -> Q, z=1 -> K.
// =====================================================================
__global__ __launch_bounds__(256) void qk_proj_kernel(
    const float* __restrict__ x,
    const float* __restrict__ Wq, const float* __restrict__ bq,
    const float* __restrict__ Wk, const float* __restrict__ bk,
    unsigned short* __restrict__ Qb, unsigned short* __restrict__ Kb)
{
  const float* W    = (blockIdx.z == 0) ? Wq : Wk;
  const float* bias = (blockIdx.z == 0) ? bq : bk;
  unsigned short* Out = (blockIdx.z == 0) ? Qb : Kb;

  const int lane = threadIdx.x & 31;
  const int wv   = threadIdx.x >> 5;
  const int m0   = blockIdx.y * 128 + wv * 16;
  const int n0   = blockIdx.x * 64;
  const int mrow = lane & 15;
  const int h    = lane >> 4;

  v8f acc[4];
#pragma unroll
  for (int t = 0; t < 4; ++t) acc[t] = zero8();

  const float* xrow = x + (size_t)(m0 + mrow) * FDIM;

  for (int k0 = 0; k0 < FDIM; k0 += 32) {
    // A fragment: rows of x, K-pairs per ISA layout -> 4 contiguous float4 loads
    FragBF A;
    const float4 fa = *(const float4*)(xrow + k0 + (h << 3));
    const float4 fb = *(const float4*)(xrow + k0 + (h << 3) + 4);
    const float4 fc = *(const float4*)(xrow + k0 + 16 + (h << 3));
    const float4 fd = *(const float4*)(xrow + k0 + 16 + (h << 3) + 4);
    A.u[0] = pack2bf(fa.x, fa.y); A.u[1] = pack2bf(fa.z, fa.w);
    A.u[2] = pack2bf(fb.x, fb.y); A.u[3] = pack2bf(fb.z, fb.w);
    A.u[4] = pack2bf(fc.x, fc.y); A.u[5] = pack2bf(fc.z, fc.w);
    A.u[6] = pack2bf(fd.x, fd.y); A.u[7] = pack2bf(fd.z, fd.w);

#pragma unroll
    for (int t = 0; t < 4; ++t) {
      const int nn = n0 + t * 16 + mrow;                          // B lane column
      FragBF B;
#pragma unroll
      for (int v = 0; v < 8; ++v) {
        const int k = (h << 3) + ((v & 3) << 1) + ((v >> 2) << 4); // B layout K-pair
        const float* p = W + (size_t)(k0 + k) * HDIM + nn;
        B.u[v] = pack2bf(p[0], p[HDIM]);
      }
      acc[t] = __builtin_amdgcn_wmma_f32_16x16x32_bf16(
          false, A.v, false, B.v, (short)0, acc[t], false, false);
    }
  }

#pragma unroll
  for (int t = 0; t < 4; ++t) {
    const int nn = n0 + t * 16 + mrow;
    const float bb = bias[nn];
#pragma unroll
    for (int v = 0; v < 8; ++v) {
      const int m = m0 + v + (h << 3);                            // C/D row = v + 8h
      Out[(size_t)m * HDIM + nn] = f2bf(acc[t][v] + bb);
    }
  }
}

// =====================================================================
// Score tiles: wave computes TWO 16x16 tiles of Q @ K^T (rows m0w and
// m0w+16) sharing one B fragment -> two independent WMMA chains.
// All fragment loads are 128-bit (2x global b128 per A, 2x ds b128 for B).
// =====================================================================
__device__ __forceinline__ void score_tile2(const unsigned* __restrict__ Qu, int m0w,
                                            const unsigned* __restrict__ Ks, int lane,
                                            v8f& c0, v8f& c1)
{
  const int mrow = lane & 15;
  const int h    = lane >> 4;
  const unsigned* q0 = Qu + (size_t)(m0w + mrow) * (HDIM / 2);
  const unsigned* q1 = q0 + 16 * (HDIM / 2);
  const unsigned* kr = Ks + mrow * (HDIM / 2);
  c0 = zero8();
  c1 = zero8();
#pragma unroll 2
  for (int k2 = 0; k2 < HDIM / 2; k2 += 16) {       // 32 k-steps of 32
    FragBF A0, A1, B;
    A0.q[0] = *(const uint4*)(q0 + k2 + (h << 2));
    A0.q[1] = *(const uint4*)(q0 + k2 + 8 + (h << 2));
    A1.q[0] = *(const uint4*)(q1 + k2 + (h << 2));
    A1.q[1] = *(const uint4*)(q1 + k2 + 8 + (h << 2));
    B.q[0]  = *(const uint4*)(kr + k2 + (h << 2));
    B.q[1]  = *(const uint4*)(kr + k2 + 8 + (h << 2));
    c0 = __builtin_amdgcn_wmma_f32_16x16x32_bf16(
        false, A0.v, false, B.v, (short)0, c0, false, false);
    c1 = __builtin_amdgcn_wmma_f32_16x16x32_bf16(
        false, A1.v, false, B.v, (short)0, c1, false, false);
  }
}

// =====================================================================
// Pass 1: online row max / row sum-of-exp over a column chunk.
// Double-buffered async K staging. Grid: (N/256, NCHUNK). Wave: 32 rows.
// =====================================================================
__global__ __launch_bounds__(256) void attn_pass1_kernel(
    const unsigned* __restrict__ Qb, const unsigned* __restrict__ Kb,
    float* __restrict__ rowMp, float* __restrict__ rowSp)
{
  __shared__ __align__(16) unsigned Ks[2][16 * (HDIM / 2)];  // 2 x 32KB K tiles
  const int lane = threadIdx.x & 31;
  const int wv   = threadIdx.x >> 5;
  const int m0w  = blockIdx.x * 256 + wv * 32;
  const int h    = lane >> 4;
  const int nbeg = blockIdx.y * CHUNKC;

  float M0[8], S0[8], M1[8], S1[8];
#pragma unroll
  for (int v = 0; v < 8; ++v) {
    M0[v] = -1e30f; S0[v] = 0.0f;
    M1[v] = -1e30f; S1[v] = 0.0f;
  }

  async_tile_load(Kb + (size_t)nbeg * (HDIM / 2), Ks[0]);

  for (int t = 0; t < NTILE; ++t) {
    const int n0 = nbeg + t * 16;
    const int p  = t & 1;
    wait_async0();            // this wave's tile-t portion landed in LDS
    __syncthreads();          // all waves' portions landed; buf p^1 free
    if (t + 1 < NTILE)
      async_tile_load(Kb + (size_t)(n0 + 16) * (HDIM / 2), Ks[p ^ 1]);

    v8f c0, c1;
    score_tile2(Qb, m0w, Ks[p], lane, c0, c1);
#pragma unroll
    for (int v = 0; v < 8; ++v) {
      {
        const float s  = c0[v] * 0.03125f;              // 1/sqrt(1024)
        const float tm = rmax16(s);
        const float nm = fmaxf(M0[v], tm);
        const float ts = rsum16(__expf(s - nm));
        S0[v] = S0[v] * __expf(M0[v] - nm) + ts;
        M0[v] = nm;
      }
      {
        const float s  = c1[v] * 0.03125f;
        const float tm = rmax16(s);
        const float nm = fmaxf(M1[v], tm);
        const float ts = rsum16(__expf(s - nm));
        S1[v] = S1[v] * __expf(M1[v] - nm) + ts;
        M1[v] = nm;
      }
    }
  }
  if ((lane & 15) == 0) {
    float* Mo = rowMp + (size_t)blockIdx.y * N_ROWS;
    float* So = rowSp + (size_t)blockIdx.y * N_ROWS;
#pragma unroll
    for (int v = 0; v < 8; ++v) {
      Mo[m0w + v + (h << 3)]      = M0[v];
      So[m0w + v + (h << 3)]      = S0[v];
      Mo[m0w + 16 + v + (h << 3)] = M1[v];
      So[m0w + 16 + v + (h << 3)] = S1[v];
    }
  }
}

// Merge per-chunk (max, sum) into global row statistics. Grid 32x256.
__global__ __launch_bounds__(256) void row_merge_kernel(
    const float* __restrict__ rowMp, const float* __restrict__ rowSp,
    float* __restrict__ rowM, float* __restrict__ rowS)
{
  const int r = blockIdx.x * 256 + threadIdx.x;
  float M = -1e30f;
#pragma unroll
  for (int c = 0; c < NCHUNK; ++c) M = fmaxf(M, rowMp[(size_t)c * N_ROWS + r]);
  float S = 0.0f;
#pragma unroll
  for (int c = 0; c < NCHUNK; ++c)
    S += rowSp[(size_t)c * N_ROWS + r] * __expf(rowMp[(size_t)c * N_ROWS + r] - M);
  rowM[r] = M;
  rowS[r] = S;
}

// =====================================================================
// Pass 2: normalized attention column sums -> per-m-block partials.
// colpart[b][j] = sum over block b's 256 rows of attn[n,j]. Deterministic.
// Double-buffered async K staging. Grid: (N/256, NCHUNK).
// =====================================================================
__global__ __launch_bounds__(256) void attn_pass2_kernel(
    const unsigned* __restrict__ Qb, const unsigned* __restrict__ Kb,
    const float* __restrict__ rowM, const float* __restrict__ rowS,
    float* __restrict__ colpart)
{
  __shared__ __align__(16) unsigned Ks[2][16 * (HDIM / 2)];
  __shared__ float colbuf[8][16];
  const int lane = threadIdx.x & 31;
  const int wv   = threadIdx.x >> 5;
  const int m0w  = blockIdx.x * 256 + wv * 32;
  const int h    = lane >> 4;
  const int nbeg = blockIdx.y * CHUNKC;

  float Mr0[8], iS0[8], Mr1[8], iS1[8];
#pragma unroll
  for (int v = 0; v < 8; ++v) {
    Mr0[v] = rowM[m0w + v + (h << 3)];
    iS0[v] = 1.0f / rowS[m0w + v + (h << 3)];
    Mr1[v] = rowM[m0w + 16 + v + (h << 3)];
    iS1[v] = 1.0f / rowS[m0w + 16 + v + (h << 3)];
  }

  async_tile_load(Kb + (size_t)nbeg * (HDIM / 2), Ks[0]);

  for (int t = 0; t < NTILE; ++t) {
    const int n0 = nbeg + t * 16;
    const int p  = t & 1;
    wait_async0();
    __syncthreads();          // tile t visible; buf p^1 and colbuf reusable
    if (t + 1 < NTILE)
      async_tile_load(Kb + (size_t)(n0 + 16) * (HDIM / 2), Ks[p ^ 1]);

    v8f c0, c1;
    score_tile2(Qb, m0w, Ks[p], lane, c0, c1);
    float partial = 0.0f;
#pragma unroll
    for (int v = 0; v < 8; ++v) {
      partial += __expf(c0[v] * 0.03125f - Mr0[v]) * iS0[v];
      partial += __expf(c1[v] * 0.03125f - Mr1[v]) * iS1[v];
    }
    const float tot = partial + __shfl_xor(partial, 16, 32);  // both row-halves
    if (lane < 16) colbuf[wv][lane] = tot;
    __syncthreads();
    if (threadIdx.x < 16) {
      float s = 0.0f;
#pragma unroll
      for (int w = 0; w < 8; ++w) s += colbuf[w][threadIdx.x];
      colpart[(size_t)blockIdx.x * N_ROWS + n0 + threadIdx.x] = s;
    }
  }
}

// colmean[j] = (1/N) * sum_b colpart[b][j].  Grid 32x256.
__global__ __launch_bounds__(256) void col_reduce_kernel(
    const float* __restrict__ colpart, float* __restrict__ w)
{
  const int j = blockIdx.x * 256 + threadIdx.x;
  float s = 0.0f;
  for (int b = 0; b < MBLK; ++b) s += colpart[(size_t)b * N_ROWS + j];
  w[j] = s * (1.0f / (float)N_ROWS);
}

// tpart[b][f] = sum over 128-row chunk of w[j] * x[j,f].  Grid 64x256.
__global__ __launch_bounds__(256) void wx_kernel(
    const float* __restrict__ w, const float* __restrict__ x,
    float* __restrict__ tpart)
{
  const int j0 = blockIdx.x * 128;
  float acc[4] = {0.0f, 0.0f, 0.0f, 0.0f};
  for (int j = 0; j < 128; ++j) {
    const float wj = w[j0 + j];
    const float* xr = x + (size_t)(j0 + j) * FDIM;
#pragma unroll
    for (int t = 0; t < 4; ++t) acc[t] += wj * xr[threadIdx.x + 256 * t];
  }
#pragma unroll
  for (int t = 0; t < 4; ++t)
    tpart[(size_t)blockIdx.x * FDIM + threadIdx.x + 256 * t] = acc[t];
}

// tvec[f] = sum_b tpart[b][f].  Grid 4x256.
__global__ __launch_bounds__(256) void t_reduce_kernel(
    const float* __restrict__ tpart, float* __restrict__ tvec)
{
  const int f = blockIdx.x * 256 + threadIdx.x;
  float s = 0.0f;
  for (int b = 0; b < 64; ++b) s += tpart[(size_t)b * FDIM + f];
  tvec[f] = s;
}

// out[f] = bv[f] + sum_f' tvec[f'] * Wv[f',f].  Grid 4x256.
__global__ __launch_bounds__(256) void out_kernel(
    const float* __restrict__ tvec, const float* __restrict__ Wv,
    const float* __restrict__ bv, float* __restrict__ out)
{
  const int f = blockIdx.x * 256 + threadIdx.x;
  float acc = bv[f];
  for (int fp = 0; fp < FDIM; ++fp) acc += tvec[fp] * Wv[(size_t)fp * FDIM + f];
  out[f] = acc;
}

extern "C" void kernel_launch(void* const* d_in, const int* in_sizes, int n_in,
                              void* d_out, int out_size, void* d_ws, size_t ws_size,
                              hipStream_t stream)
{
  const float* x  = (const float*)d_in[0];
  const float* Wq = (const float*)d_in[1];
  const float* bq = (const float*)d_in[2];
  const float* Wk = (const float*)d_in[3];
  const float* bk = (const float*)d_in[4];
  const float* Wv = (const float*)d_in[5];
  const float* bv = (const float*)d_in[6];
  float* out = (float*)d_out;

  char* ws = (char*)d_ws;
  const size_t oQ   = 0;                                      // Q bf16  16MB
  const size_t oK   = oQ   + (size_t)N_ROWS * HDIM * 2;       // K bf16  16MB
  const size_t oRMp = oK   + (size_t)N_ROWS * HDIM * 2;       // chunk row max 128KB
  const size_t oRSp = oRMp + (size_t)NCHUNK * N_ROWS * 4;     // chunk row sum 128KB
  const size_t oRM  = oRSp + (size_t)NCHUNK * N_ROWS * 4;     // row max 32KB
  const size_t oRS  = oRM  + (size_t)N_ROWS * 4;              // row sum 32KB
  const size_t oCP  = oRS  + (size_t)N_ROWS * 4;              // col partials 1MB
  const size_t oCS  = oCP  + (size_t)MBLK * N_ROWS * 4;       // colmean 32KB
  const size_t oTP  = oCS  + (size_t)N_ROWS * 4;              // t partials 256KB
  const size_t oTV  = oTP  + (size_t)64 * FDIM * 4;           // tvec 4KB

  unsigned short* Qb = (unsigned short*)(ws + oQ);
  unsigned short* Kb = (unsigned short*)(ws + oK);
  float* rowMp   = (float*)(ws + oRMp);
  float* rowSp   = (float*)(ws + oRSp);
  float* rowM    = (float*)(ws + oRM);
  float* rowS    = (float*)(ws + oRS);
  float* colpart = (float*)(ws + oCP);
  float* colmean = (float*)(ws + oCS);
  float* tpart   = (float*)(ws + oTP);
  float* tvec    = (float*)(ws + oTV);

  qk_proj_kernel<<<dim3(HDIM / 64, N_ROWS / 128, 2), 256, 0, stream>>>(
      x, Wq, bq, Wk, bk, Qb, Kb);
  attn_pass1_kernel<<<dim3(MBLK, NCHUNK), 256, 0, stream>>>(
      (const unsigned*)Qb, (const unsigned*)Kb, rowMp, rowSp);
  row_merge_kernel<<<dim3(N_ROWS / 256), 256, 0, stream>>>(rowMp, rowSp, rowM, rowS);
  attn_pass2_kernel<<<dim3(MBLK, NCHUNK), 256, 0, stream>>>(
      (const unsigned*)Qb, (const unsigned*)Kb, rowM, rowS, colpart);
  col_reduce_kernel<<<dim3(N_ROWS / 256), 256, 0, stream>>>(colpart, colmean);
  wx_kernel<<<dim3(N_ROWS / 128), 256, 0, stream>>>(colmean, x, tpart);
  t_reduce_kernel<<<dim3(FDIM / 256), 256, 0, stream>>>(tpart, tvec);
  out_kernel<<<dim3(FDIM / 256), 256, 0, stream>>>(tvec, Wv, bv, out);
}
